// NeuroScribe_77335181132256
// MI455X (gfx1250) — compile-verified, matrix operand order checked by reading, not run on hardware
//
#include <hip/hip_runtime.h>

typedef __attribute__((ext_vector_type(16))) _Float16 v16h;
typedef __attribute__((ext_vector_type(8)))  _Float16 v8h;
typedef __attribute__((ext_vector_type(8)))  float    v8f;
typedef __attribute__((ext_vector_type(4)))  unsigned int v4u;
typedef __attribute__((ext_vector_type(4)))  int      v4i;
typedef __attribute__((ext_vector_type(8)))  int      v8i;

// ---------------------------------------------------------------------------
// WMMA helpers (CDNA5 gfx1250, wave32)
// A layout (16-bit, 16x32): lane L holds row m = L%16; element i holds
//   k = (i/8)*16 + (L/16)*8 + (i%8)  -> two contiguous 8-half (16B) chunks.
// B symmetric with N in place of M -> computes A(MxK) * W(NxK)^T.
// ---------------------------------------------------------------------------
__device__ __forceinline__ v16h ld_tile16(const _Float16* __restrict__ p) {
  v8h lo = *(const v8h*)(p);
  v8h hi = *(const v8h*)(p + 16);
  return __builtin_shufflevector(lo, hi, 0,1,2,3,4,5,6,7,8,9,10,11,12,13,14,15);
}

__device__ __forceinline__ v8f wmma_f16f32(v16h a, v16h b, v8f c) {
  return __builtin_amdgcn_wmma_f32_16x16x32_f16(
      false, a, false, b, (short)0, c, false, false);
}

// ---------------------------------------------------------------------------
// TDM: 2D tile load global->LDS (f16 elements).
//   tile: rows x rowlen elems, row-major, dim0 stride = rowlen (packed)
// D# built per CDNA5 ISA. Wave-level op; caller
// must s_wait_tensorcnt + barrier before consumers read LDS.
// ---------------------------------------------------------------------------
__device__ __forceinline__ void tdm_load_2d_f16(
    unsigned lds_byte_addr, const void* gaddr,
    unsigned rows, unsigned rowlen, unsigned tensor_rows)
{
  unsigned long long ga = (unsigned long long)gaddr;
  v4u g0;
  g0[0] = 1u;                                   // count=1, user mode
  g0[1] = lds_byte_addr;                        // lds_addr [63:32]
  g0[2] = (unsigned)(ga & 0xffffffffu);         // global_addr[31:0]
  g0[3] = (unsigned)((ga >> 32) & 0x01ffffffu)  // global_addr[56:32]
        | (2u << 30);                           // type=2 ("image")
  v8i g1;
  g1[0] = (int)(1u << 16);                      // wg_mask=0, data_size=1 (2B)
  g1[1] = (int)(rowlen << 16);                  // tensor_dim0[15:0] @ bits 79:48
  g1[2] = (int)(tensor_rows << 16);             // td0 hi=0 | tensor_dim1[15:0]
  g1[3] = (int)(rowlen << 16);                  // td1 hi=0 | tile_dim0
  g1[4] = (int)rows;                            // tile_dim1 | tile_dim2=0
  g1[5] = (int)rowlen;                          // tensor_dim0_stride[31:0]
  g1[6] = 0;                                    // stride hi | dim1_stride lo
  g1[7] = 0;
  v4i z4 = {0, 0, 0, 0};
#if __clang_major__ >= 23
  v8i z8 = {0, 0, 0, 0, 0, 0, 0, 0};
  __builtin_amdgcn_tensor_load_to_lds(g0, g1, z4, z4, z8, 0);
#else
  __builtin_amdgcn_tensor_load_to_lds(g0, g1, z4, z4, 0);
#endif
}

// ---------------------------------------------------------------------------
// Kernel 0: weight prep.
//   w2t  [o][dx*64+c] f16  <- conv2_w (128,64,5) f32   (K-contig implicit GEMM)
//   fcwp [n][k] f16, n padded to 416 <- fc_w (404,3072) f32
// ---------------------------------------------------------------------------
__global__ __launch_bounds__(256) void k0_prep(
    const float* __restrict__ conv2_w, const float* __restrict__ fc_w,
    _Float16* __restrict__ w2t, _Float16* __restrict__ fcwp)
{
  size_t i = (size_t)blockIdx.x * 256 + threadIdx.x;
  if (i < (size_t)128 * 320) {
    int o = (int)(i / 320), r = (int)(i % 320);
    int dx = r >> 6, c = r & 63;
    w2t[i] = (_Float16)conv2_w[o * 320 + c * 5 + dx];
  }
  if (i < (size_t)416 * 3072) {
    int n = (int)(i / 3072), k = (int)(i % 3072);
    fcwp[i] = (n < 404) ? (_Float16)fc_w[(size_t)n * 3072 + k] : (_Float16)0.f;
  }
}

// ---------------------------------------------------------------------------
// Kernel 1: conv1 (pad 3) + ReLU + avgpool4, fp32 VALU, LDS-staged operands.
// Output h1t: f16 [b][pos+2][c], 2 zero-pad positions each side (100 x 64).
// ---------------------------------------------------------------------------
__global__ __launch_bounds__(256) void k1_conv1(
    const float* __restrict__ in, const float* __restrict__ w,
    const float* __restrict__ bias, _Float16* __restrict__ h1t)
{
  __shared__ float lin[5 * 392];
  __shared__ float lw[64 * 35];
  __shared__ float lb[64];
  const int b = blockIdx.x;
  const int t = threadIdx.x;

  for (int i = t; i < 64 * 35; i += 256) lw[i] = w[i];
  if (t < 64) lb[t] = bias[t];
  for (int i = t; i < 5 * 392; i += 256) lin[i] = 0.f;
  __syncthreads();
  for (int i = t; i < 5 * 384; i += 256) {
    int ic = i / 384, x = i % 384;
    lin[ic * 392 + 3 + x] = in[(size_t)b * 1920 + i];
  }
  __syncthreads();

  const int c = t & 63;
  const int g = t >> 6;
  const float* wc = &lw[c * 35];
  const float bc = lb[c];
  for (int pp = g * 24; pp < g * 24 + 24; ++pp) {
    float acc = 0.f;
    #pragma unroll
    for (int q = 0; q < 4; ++q) {
      const int x = pp * 4 + q;
      float s = bc;
      #pragma unroll
      for (int ic = 0; ic < 5; ++ic) {
        const float* li = &lin[ic * 392 + x];
        #pragma unroll
        for (int dx = 0; dx < 7; ++dx) s = fmaf(li[dx], wc[ic * 7 + dx], s);
      }
      acc += fmaxf(s, 0.f);
    }
    h1t[((size_t)b * 100 + pp + 2) * 64 + c] = (_Float16)(acc * 0.25f);
  }
  {
    int which = t >> 6;
    int pos = (which < 2) ? which : (96 + which);   // 0,1,98,99
    h1t[((size_t)b * 100 + pos) * 64 + (t & 63)] = (_Float16)0.f;
  }
}

// ---------------------------------------------------------------------------
// Kernel 2: conv2 as implicit GEMM via WMMA + bias + ReLU + pool4.
//   M=B*96 (16 consecutive positions of one b per tile), N=128, K=320.
// Weights staged to LDS by the Tensor Data Mover (wave 0 issues the DMA,
// s_wait_tensorcnt, workgroup barrier releases the other 7 waves).
// ---------------------------------------------------------------------------
__global__ __launch_bounds__(256) void k2_conv2(
    const _Float16* __restrict__ h1t, const _Float16* __restrict__ w2t,
    const float* __restrict__ bias2, _Float16* __restrict__ feat)
{
  __shared__ __align__(16) _Float16 sW[64 * 320];   // 40 KB
  const int t    = threadIdx.x;
  const int lane = t & 31;
  const int wid  = t >> 5;
  const int mt   = blockIdx.x * 8 + wid;
  const int b    = mt / 6;
  const int p0   = (mt % 6) * 16;
  const int lr   = lane & 15;
  const int lh   = lane >> 4;

  const _Float16* aRow = h1t + ((size_t)b * 100 + p0 + lr) * 64;  // 320 contig halves

  for (int grp = 0; grp < 2; ++grp) {
    __syncthreads();                       // previous-group readers done
    if (wid == 0) {                        // wave-uniform: TDM is a wave-level op
      tdm_load_2d_f16((unsigned)(size_t)(void*)sW,
                      w2t + (size_t)grp * 64 * 320,
                      /*rows=*/64, /*rowlen=*/320, /*tensor_rows=*/128);
      __builtin_amdgcn_s_wait_tensorcnt(0);
    }
    __syncthreads();                       // LDS tile visible to all waves

    #pragma unroll
    for (int ntl = 0; ntl < 4; ++ntl) {
      const int o = grp * 64 + ntl * 16 + lr;
      const _Float16* bRow = sW + (ntl * 16 + lr) * 320;
      v8f acc = {};
      #pragma unroll
      for (int kb = 0; kb < 10; ++kb) {
        const int k0 = kb * 32 + lh * 8;
        acc = wmma_f16f32(ld_tile16(aRow + k0), ld_tile16(bRow + k0), acc);
      }
      const float bo = bias2[o];
      float v0 = fmaxf(acc[0] + bo, 0.f), v1 = fmaxf(acc[1] + bo, 0.f);
      float v2 = fmaxf(acc[2] + bo, 0.f), v3 = fmaxf(acc[3] + bo, 0.f);
      float v4 = fmaxf(acc[4] + bo, 0.f), v5 = fmaxf(acc[5] + bo, 0.f);
      float v6 = fmaxf(acc[6] + bo, 0.f), v7 = fmaxf(acc[7] + bo, 0.f);
      const float pA = (v0 + v1 + v2 + v3) * 0.25f;
      const float pB = (v4 + v5 + v6 + v7) * 0.25f;
      const int ppb = (p0 >> 2) + lh * 2;
      _Float16* fp = feat + (size_t)b * 3072 + o * 24;
      fp[ppb]     = (_Float16)pA;
      fp[ppb + 1] = (_Float16)pB;
    }
  }
}

// ---------------------------------------------------------------------------
// Kernel 3: FC GEMM via WMMA: out(B,404) = feat(B,3072) @ fcwp(416,3072)^T + b
// Each wave: one M-tile x TWO N-tiles (shared A registers, 2 accumulators),
// k-outer. 26 n-tiles -> 13 groups.
// ---------------------------------------------------------------------------
__global__ __launch_bounds__(256) void k3_fc(
    const _Float16* __restrict__ feat, const _Float16* __restrict__ fcwp,
    const float* __restrict__ fc_b, float* __restrict__ outbuf, int nWaves)
{
  const int t    = threadIdx.x;
  const int lane = t & 31;
  const int wid  = blockIdx.x * 8 + (t >> 5);
  if (wid >= nWaves) return;               // wave-uniform; no barriers below
  const int mt = wid / 13;
  const int ng = wid % 13;                 // n-tile pair
  const int lr = lane & 15;
  const int lh = lane >> 4;

  const _Float16* aRow = feat + ((size_t)(mt * 16 + lr)) * 3072;
  const _Float16* bRow0 = fcwp + ((size_t)((ng * 2 + 0) * 16 + lr)) * 3072;
  const _Float16* bRow1 = fcwp + ((size_t)((ng * 2 + 1) * 16 + lr)) * 3072;
  v8f acc0 = {}, acc1 = {};
  for (int kb = 0; kb < 96; ++kb) {
    const int k0 = kb * 32 + lh * 8;
    v16h a = ld_tile16(aRow + k0);
    acc0 = wmma_f16f32(a, ld_tile16(bRow0 + k0), acc0);
    acc1 = wmma_f16f32(a, ld_tile16(bRow1 + k0), acc1);
  }
  #pragma unroll
  for (int half = 0; half < 2; ++half) {
    const int n = (ng * 2 + half) * 16 + lr;
    if (n < 404) {
      const float bn = fc_b[n];
      const v8f& acc = half ? acc1 : acc0;
      #pragma unroll
      for (int e = 0; e < 8; ++e) {
        const int m = mt * 16 + e + 8 * lh;
        outbuf[(size_t)m * 404 + n] = acc[e] + bn;
      }
    }
  }
}

// ---------------------------------------------------------------------------
// Kernel 4: linear head + 100-step DMP integration (thread per batch).
// out layout: y_traj (B, 101, 1) -> out[b*101 + t].
// ---------------------------------------------------------------------------
__global__ __launch_bounds__(256) void k4_dmp(
    const float* __restrict__ outbuf, const float* __restrict__ y0arr,
    const float* __restrict__ L_w, const float* __restrict__ L_b,
    float* __restrict__ out, int B)
{
  const int b = blockIdx.x * 256 + threadIdx.x;
  if (b >= B) return;
  const float* row = outbuf + (size_t)b * 404;

  float s = L_b[0];
  for (int j = 0; j < 404; ++j) s = fmaf(row[j], L_w[j], s);
  const float tau  = s;
  const float goal = row[0];
  const float w0 = row[1], w1 = row[2], w2 = row[3], w3 = row[4], w4 = row[5];
  const float y0 = y0arr[b];

  float C[5], S2[5];
  #pragma unroll
  for (int i = 0; i < 5; ++i) {
    C[i]  = __expf(-0.25f * (float)i);            // exp(-A_X * i/(N-1))
    S2[i] = 11.180339887498949f / C[i];           // 5^1.5 / C / A_X
  }
  const float DT = 0.01f;
  float x = 1.f, y = y0, z = 0.01f * tau;         // z0 = dy0 * tau
  const float scale = goal - y0;
  out[(size_t)b * 101] = y0;

  for (int tstep = 1; tstep <= 100; ++tstep) {
    x = x - x * tau * DT;
    float p0 = __expf(-0.5f * (x - C[0]) * (x - C[0]) / S2[0]);
    float p1 = __expf(-0.5f * (x - C[1]) * (x - C[1]) / S2[1]);
    float p2 = __expf(-0.5f * (x - C[2]) * (x - C[2]) / S2[2]);
    float p3 = __expf(-0.5f * (x - C[3]) * (x - C[3]) / S2[3]);
    float p4 = __expf(-0.5f * (x - C[4]) * (x - C[4]) / S2[4]);
    const float psum = p0 + p1 + p2 + p3 + p4;
    const float fx = (w0*p0 + w1*p1 + w2*p2 + w3*p3 + w4*p4) * x * scale / psum;
    const float dz = 25.f * (6.25f * (goal - y) - z) + fx;   // A_Z=25, B_Z=A_Z/4
    const float dy = z;
    y = y + dy * tau * DT;
    z = z + dz * tau * DT;
    out[(size_t)b * 101 + tstep] = y;
  }
}

// ---------------------------------------------------------------------------
// Launcher. Workspace layout (bytes, all 128B aligned), total ~87 MB:
//   h1t f16 B*100*64 | w2t f16 128*320 | fcwp f16 416*3072 |
//   feat f16 B*3072  | ob f32 B*404
// ---------------------------------------------------------------------------
extern "C" void kernel_launch(void* const* d_in, const int* in_sizes, int n_in,
                              void* d_out, int out_size, void* d_ws, size_t ws_size,
                              hipStream_t stream) {
  const float* input   = (const float*)d_in[0];
  const float* y0      = (const float*)d_in[1];
  const float* conv1_w = (const float*)d_in[2];
  const float* conv1_b = (const float*)d_in[3];
  const float* conv2_w = (const float*)d_in[4];
  const float* conv2_b = (const float*)d_in[5];
  const float* fc_w    = (const float*)d_in[6];
  const float* fc_b    = (const float*)d_in[7];
  const float* L_w     = (const float*)d_in[8];
  const float* L_b     = (const float*)d_in[9];
  float* out = (float*)d_out;

  const int B = in_sizes[1];                      // 4096

  char* ws = (char*)d_ws;
  _Float16* h1t  = (_Float16*)(ws);
  _Float16* w2t  = (_Float16*)(ws + (size_t)B * 100 * 64 * 2);
  _Float16* fcwp = (_Float16*)((char*)w2t + 128 * 320 * 2);
  _Float16* feat = (_Float16*)((char*)fcwp + (size_t)416 * 3072 * 2);
  float*    ob   = (float*)   ((char*)feat + (size_t)B * 3072 * 2);

  k0_prep<<<(416 * 3072 + 255) / 256, 256, 0, stream>>>(conv2_w, fc_w, w2t, fcwp);
  k1_conv1<<<B, 256, 0, stream>>>(input, conv1_w, conv1_b, h1t);
  k2_conv2<<<(B * 6) / 8, 256, 0, stream>>>(h1t, w2t, conv2_b, feat);
  const int nWaves = (B / 16) * 13;
  k3_fc<<<(nWaves + 7) / 8, 256, 0, stream>>>(feat, fcwp, fc_b, ob, nWaves);
  k4_dmp<<<(B + 255) / 256, 256, 0, stream>>>(ob, y0, L_w, L_b, out, B);
}
